// DecoderOnlyTransformer_54185307406593
// MI455X (gfx1250) — matrix operand order repaired and checked by reading, not compiled, not run on hardware
//
#include <hip/hip_runtime.h>
#include <hip/hip_bf16.h>
#include <stdint.h>

typedef __attribute__((ext_vector_type(16))) _Float16 v16h;
typedef __attribute__((ext_vector_type(8)))  _Float16 v8h;
typedef __attribute__((ext_vector_type(8)))  float    v8f;
typedef __attribute__((ext_vector_type(4)))  float    v4f;

#define S_LEN   129
#define MPAD    144      // 9 * 16 row tiles
#define KEYPAD  160      // 5 * 32 key padding (K dim of attn@V)
#define DM      256
#define DFF     256
#define NL      8
#define NH      4
#define DHD     64
#define STEPS   128
#define NSUB    50
#define NWAVES  32
#define TILE_LD 20       // LDS staging row stride (floats): conflict-free banks

// ---------------- workspace byte offsets (all 512B aligned) ----------------
#define OFF_WH     0u                               // 6 x [8][256][256] f16 = 6291456
#define OFF_PE     6291456u                         // [129][256] f32 = 132096
#define OFF_X32    6423552u                         // [144][256] f32 = 147456
#define OFF_TMP    6571008u                         // [144][256] f32 = 147456
#define OFF_LGT    6718464u                         // [4][144][160] f32 = 368640
#define OFF_XH     7087104u                         // [144][256] f16 = 73728
#define OFF_QH     7160832u                         // [144][256] f16 = 73728
#define OFF_KH     7234560u                         // [160][256] f16 = 81920
#define OFF_VTH    7316480u                         // [256][160] f16 (transposed V) = 81920
#define OFF_WSM    7398400u                         // [4][144][160] f16 softmax w = 184320
#define OFF_ATTNH  7582720u                         // [144][256] f16 = 73728
#define OFF_HH     7656448u                         // [144][256] f16 = 73728
#define OFF_BUF    7730176u                         // [129][64] f32 = 33024

__device__ __forceinline__ float gelu_tanh(float x) {
    float x3 = x * x * x;
    return 0.5f * x * (1.f + tanhf(0.7978845608028654f * (x + 0.044715f * x3)));
}

// epilogue modes
#define EP_F16            0
#define EP_F16_T          1
#define EP_F32            2
#define EP_F32_BIAS       3
#define EP_F16_BIAS_GELU  4

// One wave computes 16x16 tiles of C = A(MxK) * B(KxN), B[k][n] = W[n][k]
// (the x @ W.T pattern, W row-major [N][K]).
// Column-stationary mapping: each wave owns ONE output column tile, so its B
// fragments are loop-invariant registers; A loads are software-pipelined over
// K; the accumulator tile is staged through per-wave LDS so stores become one
// vectorized global_store per lane instead of 8 strided b16 stores.
template <int MODE, int KT>
__device__ void gemm_tiles(const _Float16* __restrict__ A, int lda,
                           const _Float16* __restrict__ W, int ldw,
                           void* __restrict__ Cout, int ldc,
                           const float* __restrict__ bias,
                           int Mtiles, int Ntiles,
                           int wave, int lane,
                           float* __restrict__ sh /* this wave's TILE_LD*16 buf */) {
    const int half_sel = lane >> 4;   // 0 or 1
    const int lane16   = lane & 15;
    const int wpc = NWAVES / Ntiles;          // waves per column tile
    const int col = wave / wpc;
    if (col >= Ntiles) return;                // (logits: waves 30,31 idle)
    const int n0  = col * 16;
    const int sub = wave % wpc;

    // hoist B fragments for this wave's column (KT * 32B per lane)
    v16h bfrag[KT];
    {
        const _Float16* Wrow = W + (size_t)(n0 + lane16) * ldw + 16 * half_sel;
#pragma unroll
        for (int kt = 0; kt < KT; ++kt)
            bfrag[kt] = *(const v16h*)(Wrow + kt * 32);
    }

    for (int mt = sub; mt < Mtiles; mt += wpc) {
        const int m0 = mt * 16;
        const _Float16* Arow = A + (size_t)(m0 + lane16) * lda + 16 * half_sel;
        v8f acc = {};
        // software-pipelined A loads across K tiles
        v8h lo = *(const v8h*)(Arow);
        v8h hi = *(const v8h*)(Arow + 8);
#pragma unroll
        for (int kt = 0; kt < KT; ++kt) {
            v8h lo_n, hi_n;
            if (kt + 1 < KT) {
                lo_n = *(const v8h*)(Arow + (kt + 1) * 32);
                hi_n = *(const v8h*)(Arow + (kt + 1) * 32 + 8);
            }
            v16h a;
#pragma unroll
            for (int i = 0; i < 8; ++i) { a[i] = lo[i]; a[i + 8] = hi[i]; }
            acc = __builtin_amdgcn_wmma_f32_16x16x32_f16(
                false, a, false, bfrag[kt], (short)0, acc, false, false);
            if (kt + 1 < KT) { lo = lo_n; hi = hi_n; }
        }

        // ---- stage accumulator tile into this wave's LDS buffer ----
#pragma unroll
        for (int r = 0; r < 8; ++r)
            sh[(r + 8 * half_sel) * TILE_LD + lane16] = acc[r];
        // same-wave LDS ops are hardware in-order; just stop compiler reordering
        asm volatile("" ::: "memory");

        if (MODE == EP_F16_T) {
            // lane -> V^T row n = tile col; 8 contiguous output elements
            const int n   = lane >> 1;
            const int mh8 = (lane & 1) * 8;
            v8h o;
#pragma unroll
            for (int j = 0; j < 8; ++j)
                o[j] = (_Float16)sh[(mh8 + j) * TILE_LD + n];
            _Float16* dst = (_Float16*)Cout + (size_t)(n0 + n) * ldc + m0 + mh8;
            *(v8h*)dst = o;
        } else {
            // lane -> output row m0+r2, 8 contiguous columns
            const int r2 = lane >> 1;
            const int c8 = (lane & 1) * 8;
            v4f t0 = *(const v4f*)(sh + r2 * TILE_LD + c8);
            v4f t1 = *(const v4f*)(sh + r2 * TILE_LD + c8 + 4);
            if (MODE == EP_F32) {
                float* dst = (float*)Cout + (size_t)(m0 + r2) * ldc + n0 + c8;
                *(v4f*)dst = t0; *(v4f*)(dst + 4) = t1;
            } else if (MODE == EP_F32_BIAS) {
                v4f b0 = *(const v4f*)(bias + n0 + c8);
                v4f b1 = *(const v4f*)(bias + n0 + c8 + 4);
                float* dst = (float*)Cout + (size_t)(m0 + r2) * ldc + n0 + c8;
                *(v4f*)dst = t0 + b0; *(v4f*)(dst + 4) = t1 + b1;
            } else if (MODE == EP_F16) {
                v8h o;
#pragma unroll
                for (int j = 0; j < 4; ++j) { o[j] = (_Float16)t0[j]; o[j + 4] = (_Float16)t1[j]; }
                _Float16* dst = (_Float16*)Cout + (size_t)(m0 + r2) * ldc + n0 + c8;
                *(v8h*)dst = o;
            } else {  // EP_F16_BIAS_GELU
                v4f b0 = *(const v4f*)(bias + n0 + c8);
                v4f b1 = *(const v4f*)(bias + n0 + c8 + 4);
                v8h o;
#pragma unroll
                for (int j = 0; j < 4; ++j) {
                    o[j]     = (_Float16)gelu_tanh(t0[j] + b0[j]);
                    o[j + 4] = (_Float16)gelu_tanh(t1[j] + b1[j]);
                }
                _Float16* dst = (_Float16*)Cout + (size_t)(m0 + r2) * ldc + n0 + c8;
                *(v8h*)dst = o;
            }
        }
        asm volatile("" ::: "memory");  // keep next tile's writes after reads
    }
}

// x = LN(x + addend) on rows [0, S_LEN); zero padded rows; mirror into f16.
__device__ void ln_residual(float* __restrict__ x32, _Float16* __restrict__ xh,
                            const float* __restrict__ addend,
                            const float* __restrict__ w, const float* __restrict__ b,
                            int tid) {
    const int wave = tid >> 5, lane = tid & 31;
    for (int r = wave; r < S_LEN; r += NWAVES) {
        float vals[8];
        float s = 0.f;
#pragma unroll
        for (int j = 0; j < 8; ++j) {
            const int c = lane + 32 * j;
            float v = x32[(size_t)r * DM + c] + addend[(size_t)r * DM + c];
            vals[j] = v; s += v;
        }
#pragma unroll
        for (int m = 16; m > 0; m >>= 1) s += __shfl_xor(s, m, 32);
        const float mu = s * (1.f / 256.f);
        float vs = 0.f;
#pragma unroll
        for (int j = 0; j < 8; ++j) { float d = vals[j] - mu; vs += d * d; }
#pragma unroll
        for (int m = 16; m > 0; m >>= 1) vs += __shfl_xor(vs, m, 32);
        const float rstd = rsqrtf(vs * (1.f / 256.f) + 1e-5f);
#pragma unroll
        for (int j = 0; j < 8; ++j) {
            const int c = lane + 32 * j;
            float y = (vals[j] - mu) * rstd * w[c] + b[c];
            x32[(size_t)r * DM + c] = y;
            xh[(size_t)r * DM + c] = (_Float16)y;
        }
    }
    for (int idx = tid; idx < (MPAD - S_LEN) * DM; idx += 1024) {
        const size_t p = (size_t)S_LEN * DM + idx;
        x32[p] = 0.f; xh[p] = (_Float16)0.f;
    }
}

// softmax with the reference's mask semantics: keep where q < k, masked real
// keys get finfo.min (all-masked row -> uniform over the 129 real keys),
// padded keys get weight exactly 0.
__device__ void softmax_phase(const float* __restrict__ lgt,
                              _Float16* __restrict__ wsm, int tid) {
    const int wave = tid >> 5, lane = tid & 31;
    const float FMIN = -3.402823466e38f;
    for (int job = wave; job < NH * MPAD; job += NWAVES) {
        const int h = job / MPAD, q = job % MPAD;
        const float* row = lgt + ((size_t)h * MPAD + q) * KEYPAD;
        _Float16* wrow = wsm + ((size_t)h * MPAD + q) * KEYPAD;
        float vals[5];
        float mx = FMIN;
#pragma unroll
        for (int j = 0; j < 5; ++j) {
            const int k = lane + 32 * j;
            float v = FMIN;
            if (k < S_LEN) {
                v = (q < k) ? row[k] * 0.125f : FMIN;   // 1/sqrt(DH)
                mx = fmaxf(mx, v);
            }
            vals[j] = v;
        }
#pragma unroll
        for (int m = 16; m > 0; m >>= 1) mx = fmaxf(mx, __shfl_xor(mx, m, 32));
        float es[5];
        float sum = 0.f;
#pragma unroll
        for (int j = 0; j < 5; ++j) {
            const int k = lane + 32 * j;
            float e = (k < S_LEN) ? __expf(vals[j] - mx) : 0.f;
            es[j] = e; sum += e;
        }
#pragma unroll
        for (int m = 16; m > 0; m >>= 1) sum += __shfl_xor(sum, m, 32);
        const float rs = 1.f / sum;
#pragma unroll
        for (int j = 0; j < 5; ++j) wrow[lane + 32 * j] = (_Float16)(es[j] * rs);
    }
}

// x = (buf @ emb_w.T + emb_b) * 16 + pe  (f32 VALU; K=64 matvec-ish)
__device__ void embed_phase(const float* __restrict__ buf,
                            const float* __restrict__ emb_w,
                            const float* __restrict__ emb_b,
                            const float* __restrict__ pe,
                            float* __restrict__ x32, _Float16* __restrict__ xh,
                            int tid) {
    for (int job = tid; job < S_LEN * DM; job += 1024) {
        const int s = job / DM, o = job % DM;
        const float* br = buf + (size_t)s * 64;
        const float* wr = emb_w + (size_t)o * 64;
        float acc = 0.f;
#pragma unroll 8
        for (int j = 0; j < 64; ++j) acc += br[j] * wr[j];
        const float v = (acc + emb_b[o]) * 16.f + pe[job];
        x32[job] = v; xh[job] = (_Float16)v;
    }
    for (int idx = tid; idx < (MPAD - S_LEN) * DM; idx += 1024) {
        const size_t p = (size_t)S_LEN * DM + idx;
        x32[p] = 0.f; xh[p] = (_Float16)0.f;
    }
}

__global__ void prep_weights_kernel(const float* __restrict__ Wq,
                                    const float* __restrict__ Wk,
                                    const float* __restrict__ Wv,
                                    const float* __restrict__ Wo,
                                    const float* __restrict__ W1,
                                    const float* __restrict__ W2,
                                    _Float16* __restrict__ dst) {
    const size_t per = (size_t)NL * DM * DM;      // 524288
    size_t idx = (size_t)blockIdx.x * blockDim.x + threadIdx.x;
    if (idx >= 6 * per) return;
    const int a = (int)(idx / per);
    const size_t r = idx % per;
    const float* src = (a == 0) ? Wq : (a == 1) ? Wk : (a == 2) ? Wv
                     : (a == 3) ? Wo : (a == 4) ? W1 : W2;
    dst[idx] = (_Float16)src[r];
}

__global__ __launch_bounds__(1024)
void transformer_persistent_kernel(
    const float* __restrict__ x0,
    const float* __restrict__ emb_w, const float* __restrict__ emb_b,
    const float* __restrict__ ln1w, const float* __restrict__ ln1b,
    const float* __restrict__ b1v, const float* __restrict__ b2v,
    const float* __restrict__ ln2w, const float* __restrict__ ln2b,
    const float* __restrict__ rw1, const float* __restrict__ rb1,
    const float* __restrict__ rw2, const float* __restrict__ rb2,
    float* __restrict__ out, char* __restrict__ ws) {
    const int tid = threadIdx.x;
    const int wave = tid >> 5;
    const int lane = tid & 31;

    const _Float16* WH = (const _Float16*)(ws + OFF_WH);
    const size_t per = (size_t)NL * DM * DM;
    const _Float16* WqH = WH;
    const _Float16* WkH = WH + per;
    const _Float16* WvH = WH + 2 * per;
    const _Float16* WoH = WH + 3 * per;
    const _Float16* W1H = WH + 4 * per;
    const _Float16* W2H = WH + 5 * per;

    float*      pe    = (float*)(ws + OFF_PE);
    float*      x32   = (float*)(ws + OFF_X32);
    float*      tmp32 = (float*)(ws + OFF_TMP);
    float*      lgt   = (float*)(ws + OFF_LGT);
    _Float16*   xh    = (_Float16*)(ws + OFF_XH);
    _Float16*   qh    = (_Float16*)(ws + OFF_QH);
    _Float16*   kh    = (_Float16*)(ws + OFF_KH);
    _Float16*   vth   = (_Float16*)(ws + OFF_VTH);
    _Float16*   wsm   = (_Float16*)(ws + OFF_WSM);
    _Float16*   attnh = (_Float16*)(ws + OFF_ATTNH);
    _Float16*   hh    = (_Float16*)(ws + OFF_HH);
    float*      buf   = (float*)(ws + OFF_BUF);

    __shared__ float sh_tile[NWAVES * TILE_LD * 16];  // 40KB tile staging
    __shared__ float sh_v[64];
    __shared__ float sh_we[256];
    __shared__ float sh_h[256];
    __shared__ float sh_p[1024];
    float* mytile = sh_tile + wave * (TILE_LD * 16);

    // ---------------- one-time init ----------------
    for (int job = tid; job < S_LEN * DM; job += 1024) {   // sinusoidal PE
        const int s = job / DM, c = job % DM;
        const float d2 = (float)(c & ~1);
        const float div = __expf(-9.210340371976184f * d2 * (1.f / 256.f));
        const float a = (float)s * div;
        pe[job] = (c & 1) ? cosf(a) : sinf(a);
    }
    for (int i = tid; i < KEYPAD * DM; i += 1024) kh[i]  = (_Float16)0.f;
    for (int i = tid; i < DM * KEYPAD; i += 1024) vth[i] = (_Float16)0.f;
    for (int i = tid; i < S_LEN * 64; i += 1024)  buf[i] = (i < 64) ? x0[i] : 0.f;
    __syncthreads();

    for (int step = 0; step < STEPS; ++step) {
        embed_phase(buf, emb_w, emb_b, pe, x32, xh, tid);
        __syncthreads();

        for (int l = 0; l < NL; ++l) {
            const _Float16* wq = WqH + (size_t)l * DM * DM;
            const _Float16* wk = WkH + (size_t)l * DM * DM;
            const _Float16* wv = WvH + (size_t)l * DM * DM;
            const _Float16* wo = WoH + (size_t)l * DM * DM;
            const _Float16* w1 = W1H + (size_t)l * DM * DM;
            const _Float16* w2 = W2H + (size_t)l * DM * DM;
            __builtin_prefetch(WqH + (((size_t)(l + 1)) % NL) * DM * DM + tid * 16, 0, 1);

            // Q, K, V projections (V stored transposed for the AV GEMM)
            gemm_tiles<EP_F16, 8>(xh, DM, wq, DM, qh, DM, nullptr, 9, 16, wave, lane, mytile);
            gemm_tiles<EP_F16, 8>(xh, DM, wk, DM, kh, DM, nullptr, 9, 16, wave, lane, mytile);
            gemm_tiles<EP_F16_T, 8>(xh, DM, wv, DM, vth, KEYPAD, nullptr, 9, 16, wave, lane, mytile);
            __syncthreads();

            // per-head logits: q_h (Mx64) x k_h^T (64xKEYS)
            for (int h = 0; h < NH; ++h)
                gemm_tiles<EP_F32, 2>(qh + h * DHD, DM, kh + h * DHD, DM,
                                      lgt + (size_t)h * MPAD * KEYPAD, KEYPAD,
                                      nullptr, 9, 10, wave, lane, mytile);
            __syncthreads();

            softmax_phase(lgt, wsm, tid);
            __syncthreads();

            // attn = w @ V   (B operand = V^T rows, contiguous along keys)
            for (int h = 0; h < NH; ++h)
                gemm_tiles<EP_F16, 5>(wsm + (size_t)h * MPAD * KEYPAD, KEYPAD,
                                      vth + (size_t)h * DHD * KEYPAD, KEYPAD,
                                      attnh + h * DHD, DM, nullptr, 9, 4, wave, lane, mytile);
            __syncthreads();

            // output projection, residual + LN1
            gemm_tiles<EP_F32, 8>(attnh, DM, wo, DM, tmp32, DM, nullptr, 9, 16, wave, lane, mytile);
            __syncthreads();
            ln_residual(x32, xh, tmp32, ln1w + l * DM, ln1b + l * DM, tid);
            __syncthreads();

            // MLP: gelu(x@W1.T + b1) @ W2.T + b2, residual + LN2
            gemm_tiles<EP_F16_BIAS_GELU, 8>(xh, DM, w1, DM, hh, DM, b1v + l * DFF,
                                            9, 16, wave, lane, mytile);
            __syncthreads();
            gemm_tiles<EP_F32_BIAS, 8>(hh, DM, w2, DM, tmp32, DM, b2v + l * DM,
                                       9, 16, wave, lane, mytile);
            __syncthreads();
            ln_residual(x32, xh, tmp32, ln2w + l * DM, ln2b + l * DM, tid);
            __syncthreads();
        }

        // ---------------- 50-substep refinement (serial matvecs, f32) ----------------
        const float* ctx = x32 + (size_t)step * DM;     // x[i]
        if (tid < 64) sh_v[tid] = buf[(size_t)step * 64 + tid];
        __syncthreads();

        for (int sub = 0; sub < NSUB; ++sub) {
            // w_emb = (emb_w @ v + emb_b) * 16
            if (tid < 256) {
                const float* wr = emb_w + (size_t)tid * 64;
                float acc = 0.f;
#pragma unroll 8
                for (int j = 0; j < 64; ++j) acc += wr[j] * sh_v[j];
                sh_we[tid] = (acc + emb_b[tid]) * 16.f;
            }
            __syncthreads();
            // h = gelu(ref_w1 @ [w_emb; ctx] + b1): 256 outs x dot512, 4 lanes/out
            {
                const int o = tid >> 2, part = tid & 3;
                const float* wr = rw1 + (size_t)o * 512 + part * 128;
                float p = 0.f;
#pragma unroll 8
                for (int j = 0; j < 128; ++j) {
                    const int k = part * 128 + j;
                    const float c = (k < 256) ? sh_we[k] : ctx[k - 256];
                    p += wr[j] * c;
                }
                sh_p[tid] = p;
            }
            __syncthreads();
            if (tid < 256) {
                float s = sh_p[4 * tid] + sh_p[4 * tid + 1] + sh_p[4 * tid + 2] + sh_p[4 * tid + 3];
                sh_h[tid] = gelu_tanh(s + rb1[tid]);
            }
            __syncthreads();
            // v += ref_w2 @ h + b2: 64 outs x dot256, 4 lanes/out
            if (tid < 256) {
                const int o = tid >> 2, part = tid & 3;
                const float* wr = rw2 + (size_t)o * 256 + part * 64;
                float p = 0.f;
#pragma unroll 8
                for (int j = 0; j < 64; ++j) p += wr[j] * sh_h[part * 64 + j];
                sh_p[tid] = p;
            }
            __syncthreads();
            if (tid < 64) {
                float s = sh_p[4 * tid] + sh_p[4 * tid + 1] + sh_p[4 * tid + 2] + sh_p[4 * tid + 3];
                sh_v[tid] = sh_v[tid] + s + rb2[tid];
            }
            __syncthreads();
        }

        if (tid < 64) {
            const float nv = sh_v[tid];
            out[(size_t)step * 64 + tid] = nv;          // preds[step]
            buf[(size_t)(step + 1) * 64 + tid] = nv;    // b[i+1]
        }
        __syncthreads();
    }
}

extern "C" void kernel_launch(void* const* d_in, const int* in_sizes, int n_in,
                              void* d_out, int out_size, void* d_ws, size_t ws_size,
                              hipStream_t stream) {
    (void)in_sizes; (void)n_in; (void)out_size; (void)ws_size;
    const float* x0    = (const float*)d_in[0];
    const float* emb_w = (const float*)d_in[1];
    const float* emb_b = (const float*)d_in[2];
    const float* Wq    = (const float*)d_in[3];
    const float* Wk    = (const float*)d_in[4];
    const float* Wv    = (const float*)d_in[5];
    const float* Wo    = (const float*)d_in[6];
    const float* ln1w  = (const float*)d_in[7];
    const float* ln1b  = (const float*)d_in[8];
    const float* w1    = (const float*)d_in[9];
    const float* b1    = (const float*)d_in[10];
    const float* w2    = (const float*)d_in[11];
    const float* b2    = (const float*)d_in[12];
    const float* ln2w  = (const float*)d_in[13];
    const float* ln2b  = (const float*)d_in[14];
    const float* rw1   = (const float*)d_in[15];
    const float* rb1   = (const float*)d_in[16];
    const float* rw2   = (const float*)d_in[17];
    const float* rb2   = (const float*)d_in[18];
    // d_in[19] = steps (always 128; baked in as STEPS)

    _Float16* WH = (_Float16*)d_ws;
    const int total = 6 * NL * DM * DM;
    prep_weights_kernel<<<(total + 255) / 256, 256, 0, stream>>>(Wq, Wk, Wv, Wo, w1, w2, WH);
    transformer_persistent_kernel<<<1, 1024, 0, stream>>>(
        x0, emb_w, emb_b, ln1w, ln1b, b1, b2, ln2w, ln2b,
        rw1, rb1, rw2, rb2, (float*)d_out, (char*)d_ws);
}
